// QuantLinearORT_54262616818181
// MI455X (gfx1250) — matrix operand order repaired and checked by reading, not compiled, not run on hardware
//
#include <hip/hip_runtime.h>

typedef __attribute__((ext_vector_type(16))) __bf16 v16bf;
typedef __attribute__((ext_vector_type(8)))  __bf16 v8bf;
typedef __attribute__((ext_vector_type(2)))  __bf16 v2bf;
typedef __attribute__((ext_vector_type(8)))  float  v8f;

#define IN_F   4096
#define OUT_F  4096
#define KB_CNT 32        // K blocks (groups of 128)
#define BM     128
#define BN     128
#define LDK    136       // padded bf16 stride (272 B = 17*16 -> 16B aligned rows)
#define LDKW   68        // uint stride (= LDK/2)

// packed f32 -> bf16 (RNE). Prefer the HW packed convert on CDNA5.
static __device__ __forceinline__ unsigned int f2bf_pack(float a, float b) {
#if __has_builtin(__builtin_amdgcn_cvt_pk_bf16_f32)
  v2bf r = __builtin_amdgcn_cvt_pk_bf16_f32(a, b);
  return __builtin_bit_cast(unsigned int, r);
#else
  unsigned ua = __float_as_uint(a);
  unsigned ub = __float_as_uint(b);
  ua = ua + 0x7FFFu + ((ua >> 16) & 1u);
  ub = ub + 0x7FFFu + ((ub >> 16) & 1u);
  return (ua >> 16) | (ub & 0xFFFF0000u);
#endif
}

__global__ __launch_bounds__(256) void qlin_wmma_bf16(
    const float* __restrict__ x, const int* __restrict__ qw,
    const int* __restrict__ qz, const float* __restrict__ sc,
    const float* __restrict__ bias, float* __restrict__ out)
{
  __shared__ __align__(16) unsigned short sA[BM * LDK];   // activations bf16 [m][k]
  __shared__ __align__(16) unsigned short sB[BN * LDK];   // dequant weights bf16 [n][k]
  unsigned int* sAu = (unsigned int*)sA;
  unsigned int* sBu = (unsigned int*)sB;

  const int tid  = threadIdx.x;
  const int lane = tid & 31;
  const int wid  = tid >> 5;
  const int mb   = blockIdx.y * BM;
  const int nb   = blockIdx.x * BN;

  // wave sub-tile: 2 (M) x 4 (N) waves, each owns 64x32 output
  const int wm = (wid >> 2) * 64;
  const int wn = (wid & 3) * 32;

  v8f acc[4][2];
  const v8f vzero = {0.f, 0.f, 0.f, 0.f, 0.f, 0.f, 0.f, 0.f};
#pragma unroll
  for (int i = 0; i < 4; ++i)
#pragma unroll
    for (int j = 0; j < 2; ++j)
      acc[i][j] = vzero;

  // staging assignments (2 threads per row/col)
  const int arow   = tid >> 1;         // 0..127  A row
  const int akoff  = (tid & 1) * 64;   // which 64 floats of the 128-k slab
  const int wcol   = tid >> 1;         // 0..127  W output column within tile
  const int whalf  = tid & 1;          // which 32 packed bytes of the group
  const int ncol_g = nb + wcol;

  for (int kb = 0; kb < KB_CNT; ++kb) {
    // ---------- stage A: fp32 -> bf16 into LDS ----------
    const float4* gx = (const float4*)(x + (size_t)(mb + arow) * IN_F + kb * 128 + akoff);
    unsigned int* da = sAu + arow * LDKW + (akoff >> 1);
    if (kb + 1 < KB_CNT) {
      __builtin_prefetch(x + (size_t)(mb + arow) * IN_F + (kb + 1) * 128 + akoff, 0, 3);
      __builtin_prefetch(qw + (size_t)ncol_g * (KB_CNT * 64) + (kb + 1) * 64 + whalf * 32, 0, 3);
    }
#pragma unroll
    for (int i = 0; i < 16; ++i) {
      float4 v = gx[i];
      da[2 * i + 0] = f2bf_pack(v.x, v.y);
      da[2 * i + 1] = f2bf_pack(v.z, v.w);
    }

    // ---------- stage W: dequant int4 -> bf16 into LDS (transposed [n][k]) ----------
    const int   sidx = ncol_g * KB_CNT + kb;
    const float s    = sc[sidx];
    const int   zraw = qz[sidx >> 1];
    const float nzs  = -(float)((sidx & 1) ? ((zraw >> 4) & 15) : (zraw & 15)) * s;
    const int4* gq = (const int4*)(qw + (size_t)ncol_g * (KB_CNT * 64) + kb * 64 + whalf * 32);
    unsigned int* db = sBu + wcol * LDKW + whalf * 32;
#pragma unroll
    for (int i = 0; i < 8; ++i) {
      int4 q = gq[i];  // each int holds one packed byte (two nibbles, low first)
      db[4 * i + 0] = f2bf_pack(fmaf((float)(q.x & 15), s, nzs),
                                fmaf((float)((q.x >> 4) & 15), s, nzs));
      db[4 * i + 1] = f2bf_pack(fmaf((float)(q.y & 15), s, nzs),
                                fmaf((float)((q.y >> 4) & 15), s, nzs));
      db[4 * i + 2] = f2bf_pack(fmaf((float)(q.z & 15), s, nzs),
                                fmaf((float)((q.z >> 4) & 15), s, nzs));
      db[4 * i + 3] = f2bf_pack(fmaf((float)(q.w & 15), s, nzs),
                                fmaf((float)((q.w >> 4) & 15), s, nzs));
    }
    __syncthreads();

    // ---------- WMMA over the 128-deep K slab (4 steps of K=32) ----------
    const int mrow = lane & 15;
    const int akb  = (lane >> 4) * 8;    // A: half-wave K phase (halfs)
    const int bkb  = (lane >> 4) * 8;    // B: half-wave K phase (halfs)
#pragma unroll
    for (int ks = 0; ks < 4; ++ks) {
      const int k0 = ks * 32;
      v16bf af[4];
#pragma unroll
      for (int t = 0; t < 4; ++t) {
        const unsigned short* pa = sA + (size_t)(wm + t * 16 + mrow) * LDK + k0 + akb;
        v8bf lo = *(const v8bf*)pa;
        v8bf hi = *(const v8bf*)(pa + 16);
        af[t] = __builtin_shufflevector(lo, hi, 0, 1, 2, 3, 4, 5, 6, 7,
                                                 8, 9, 10, 11, 12, 13, 14, 15);
      }
      v16bf bfr[2];
#pragma unroll
      for (int t = 0; t < 2; ++t) {
        const unsigned short* pb = sB + (size_t)(wn + t * 16 + mrow) * LDK + k0 + bkb;
        v8bf lo = *(const v8bf*)pb;
        v8bf hi = *(const v8bf*)(pb + 16);
        bfr[t] = __builtin_shufflevector(lo, hi, 0, 1, 2, 3, 4, 5, 6, 7,
                                                  8, 9, 10, 11, 12, 13, 14, 15);
      }
#pragma unroll
      for (int i = 0; i < 4; ++i)
#pragma unroll
        for (int j = 0; j < 2; ++j)
          acc[i][j] = __builtin_amdgcn_wmma_f32_16x16x32_bf16(
              false, af[i], false, bfr[j], (short)0, acc[i][j], false, false);
    }
    __syncthreads();
  }

  // ---------- epilogue: bias + non-temporal stores ----------
#pragma unroll
  for (int j = 0; j < 2; ++j) {
    const int   col = nb + wn + j * 16 + (lane & 15);
    const float bv  = bias[col];
#pragma unroll
    for (int i = 0; i < 4; ++i) {
      const int row0 = mb + wm + i * 16 + (lane >> 4) * 8;
      float* o = out + (size_t)row0 * OUT_F + col;
#pragma unroll
      for (int r = 0; r < 8; ++r)
        __builtin_nontemporal_store(acc[i][j][r] + bv, o + (size_t)r * OUT_F);
    }
  }
}

extern "C" void kernel_launch(void* const* d_in, const int* in_sizes, int n_in,
                              void* d_out, int out_size, void* d_ws, size_t ws_size,
                              hipStream_t stream) {
  const float* x    = (const float*)d_in[0];
  const int*   qw   = (const int*)d_in[1];
  const int*   qz   = (const int*)d_in[2];
  const float* sc   = (const float*)d_in[3];
  const float* bias = (const float*)d_in[4];
  float*       out  = (float*)d_out;

  const int Mrows = in_sizes[0] / IN_F;           // 4*2048 = 8192
  dim3 grid(OUT_F / BN, Mrows / BM);              // (32, 64)
  qlin_wmma_bf16<<<grid, dim3(256), 0, stream>>>(x, qw, qz, sc, bias, out);
}
